// GraphUnet_17901423690467
// MI455X (gfx1250) — compile-verified, matrix operand order checked by reading, not run on hardware
//
#include <hip/hip_runtime.h>

// ---------------------------------------------------------------------------
// Graph-U-Net forward for MI455X (gfx1250, wave32, WMMA + Tensor Data Mover).
// Heavy n*n*d products run through v_wmma_f32_16x16x32_bf16; full tiles are
// staged global->LDS by TDM (tensor_load_to_lds) with hardware LDS padding.
// Block = 256 threads = 8 waves (4 row-tiles x 2 col-groups), 64x64 C tile,
// two 16x16 fragments per wave per 32-deep K step.
// ---------------------------------------------------------------------------

typedef __attribute__((ext_vector_type(16))) __bf16    v16bf;
typedef __attribute__((ext_vector_type(8)))  float     v8f;
typedef __attribute__((ext_vector_type(4)))  unsigned  uint32x4;
typedef __attribute__((ext_vector_type(8)))  int       int32x8;
typedef __attribute__((ext_vector_type(4)))  int       int32x4;

union FragBF { v16bf v; unsigned u[8]; };

__device__ __forceinline__ unsigned short f2bf(float x) {
  unsigned u = __builtin_bit_cast(unsigned, x);
  return (unsigned short)((u + 0x7FFFu + ((u >> 16) & 1u)) >> 16);  // RNE
}
__device__ __forceinline__ unsigned pk2bf(float x0, float x1) {
  return (unsigned)f2bf(x0) | ((unsigned)f2bf(x1) << 16);
}

#define TM 64
#define TN 64
#define TK 32
#define LSAF 34  // lsAf row stride (floats): 32 data + 2 pad = 136B rows
#define LSB  68  // lsB  row stride (shorts): 64 data + 4 pad = 136B rows
#define LSP  36  // lsP  row stride (shorts): 72B rows (P tile, manual fill)

// ---------------------------------------------------------------------------
// TDM 2D tile load: global (row-major, stride0 elems/row) -> LDS, padded rows.
// Tile rows are 128B; pad_interval = 32 DWORDs (code 4), pad = 2 DWORDs (code 1)
// -> LDS row stride 136B, matching LSAF/LSB layouts.
// D# packing per CDNA5 ISA.
// ---------------------------------------------------------------------------
__device__ __forceinline__ void tdm_load_2d(unsigned ldsOff, const void* gptr,
                                            unsigned dsz_code,      // 1=2B, 2=4B
                                            unsigned tdim0, unsigned tdim1,
                                            unsigned tile0, unsigned tile1,
                                            unsigned long long stride0) {
  unsigned long long ga = (unsigned long long)gptr;
  uint32x4 g0;
  g0[0] = 1u;                                        // count=1 (valid user D#)
  g0[1] = ldsOff;                                    // lds_addr (bytes)
  g0[2] = (unsigned)(ga & 0xFFFFFFFFull);            // global_addr[31:0]
  g0[3] = (unsigned)((ga >> 32) & 0x01FFFFFFull) | 0x80000000u;  // [56:32]|type=2
  int32x8 g1;
  g1[0] = (int)((dsz_code << 16) | (1u << 20) | (4u << 22) | (1u << 25));
  g1[1] = (int)((tdim0 & 0xFFFFu) << 16);            // tensor_dim0[15:0]
  g1[2] = (int)(((tdim0 >> 16) & 0xFFFFu) | ((tdim1 & 0xFFFFu) << 16));
  g1[3] = (int)(((tdim1 >> 16) & 0xFFFFu) | ((tile0 & 0xFFFFu) << 16));
  g1[4] = (int)(tile1 & 0xFFFFu);                    // tile_dim1 (tile_dim2=0)
  g1[5] = (int)(stride0 & 0xFFFFFFFFull);            // tensor_dim0_stride[31:0]
  g1[6] = (int)((stride0 >> 32) & 0xFFFFull);        // stride[47:32]
  g1[7] = 0;
  int32x4 gz = {0, 0, 0, 0};
  int32x8 gz8 = {0, 0, 0, 0, 0, 0, 0, 0};
  __builtin_amdgcn_tensor_load_to_lds(g0, g1, gz, gz, gz8, 0);
}

// ---------------------------------------------------------------------------
// Generic GEMM: C[M,N] = A[M,K](f32) @ B[K,N](bf16) (+bias)(+resid)(+ELU)
// ---------------------------------------------------------------------------
__global__ __launch_bounds__(256)
void wmma_gemm_bf16(const float* __restrict__ A, int lda,
                    const unsigned short* __restrict__ B, int ldb,
                    float* __restrict__ C, int ldc,
                    int M, int N, int K,
                    const float* __restrict__ bias,
                    const float* __restrict__ resid,
                    int act) {
  __shared__ float          lsAf[TM * LSAF];
  __shared__ unsigned short lsB[TK * LSB];

  const int tid  = threadIdx.x;
  const int bm   = blockIdx.y * TM;
  const int bn   = blockIdx.x * TN;
  const int wid  = tid >> 5, lane = tid & 31;
  const int wm   = wid & 3,  wn   = wid >> 2;
  const int ml   = lane & 15, hf  = lane >> 4;
  const int nb0  = wn * 32;

  const bool fullM = (bm + TM <= M);
  const bool fullN = (bn + TN <= N);
  const unsigned ldsA = (unsigned)(unsigned long long)(const void*)&lsAf[0];
  const unsigned ldsB = (unsigned)(unsigned long long)(const void*)&lsB[0];

  v8f acc0 = {0.f, 0.f, 0.f, 0.f, 0.f, 0.f, 0.f, 0.f};
  v8f acc1 = {0.f, 0.f, 0.f, 0.f, 0.f, 0.f, 0.f, 0.f};

  const int Kt = (K + TK - 1) / TK;
  for (int kt = 0; kt < Kt; ++kt) {
    const int k0 = kt * TK;
    const bool fullK = (k0 + TK <= K);

    // ---- A tile: 64x32 fp32 ----
    if (fullM && fullK) {
      if (wid == 0)
        tdm_load_2d(ldsA, &A[(size_t)bm * lda + k0], /*dsz=*/2u,
                    (unsigned)K, (unsigned)M, TK, TM, (unsigned long long)lda);
    } else {
#pragma unroll
      for (int t = 0; t < 8; ++t) {            // branch-free clamp + select
        int e = tid + t * 256;
        int r = e >> 5, c = e & 31;
        int gr = bm + r, gc = k0 + c;
        bool in = (gr < M) & (gc < K);
        float x = A[in ? ((size_t)gr * lda + gc) : 0];
        lsAf[r * LSAF + c] = in ? x : 0.f;
      }
    }
    // ---- B tile: 32(K) x 64(N) bf16 ----
    if (fullN && fullK) {
      if (wid == 1)
        tdm_load_2d(ldsB, &B[(size_t)k0 * ldb + bn], /*dsz=*/1u,
                    (unsigned)N, (unsigned)K, TN, TK, (unsigned long long)ldb);
    } else {
#pragma unroll
      for (int t = 0; t < 8; ++t) {
        int e = tid + t * 256;
        int r = e >> 6, c = e & 63;
        int gr = k0 + r, gc = bn + c;
        bool in = (gr < K) & (gc < N);
        unsigned short x = B[in ? ((size_t)gr * ldb + gc) : 0];
        lsB[r * LSB + c] = in ? x : (unsigned short)0;
      }
    }
    __builtin_amdgcn_s_wait_tensorcnt((short)0);
    __syncthreads();

    FragBF fa, fb0, fb1;
#pragma unroll
    for (int q = 0; q < 8; ++q) {
      int kk = (q < 4) ? (2 * q + 8 * hf) : (16 + 2 * (q - 4) + 8 * hf);
      const float2 a2 = *(const float2*)&lsAf[(wm * 16 + ml) * LSAF + kk];
      fa.u[q] = pk2bf(a2.x, a2.y);
      unsigned lo0 = lsB[kk * LSB + nb0 + ml];
      unsigned hi0 = lsB[(kk + 1) * LSB + nb0 + ml];
      fb0.u[q] = lo0 | (hi0 << 16);
      unsigned lo1 = lsB[kk * LSB + nb0 + 16 + ml];
      unsigned hi1 = lsB[(kk + 1) * LSB + nb0 + 16 + ml];
      fb1.u[q] = lo1 | (hi1 << 16);
    }
    acc0 = __builtin_amdgcn_wmma_f32_16x16x32_bf16(false, fa.v, false, fb0.v,
                                                   (short)0, acc0, false, false);
    acc1 = __builtin_amdgcn_wmma_f32_16x16x32_bf16(false, fa.v, false, fb1.v,
                                                   (short)0, acc1, false, false);
    __syncthreads();
  }

#pragma unroll
  for (int q = 0; q < 8; ++q) {
    int row = bm + wm * 16 + q + 8 * hf;
    if (row >= M) continue;
#pragma unroll
    for (int s = 0; s < 2; ++s) {
      int col = bn + nb0 + s * 16 + ml;
      if (col < N) {
        float x = (s == 0) ? acc0[q] : acc1[q];
        if (bias)  x += bias[col];
        if (resid) x += resid[(size_t)row * ldc + col];
        if (act == 1) x = x > 0.f ? x : (__expf(x) - 1.f);   // ELU
        C[(size_t)row * ldc + col] = x;
      }
    }
  }
}

// ---------------------------------------------------------------------------
// GAT attention * values: O[n,nc] = softmax_row(mask(lrelu(es_i+ed_j))) @ H
// P tile generated on the fly -> bf16 LDS; H tile (bf16) staged by TDM.
// ---------------------------------------------------------------------------
__global__ __launch_bounds__(256)
void wmma_gat_av(const float* __restrict__ Adj, int lda, int n,
                 const float* __restrict__ es, const float* __restrict__ ed,
                 const float* __restrict__ rowm, const float* __restrict__ rsinv,
                 const unsigned short* __restrict__ H, float* __restrict__ O,
                 int nc) {
  __shared__ unsigned short lsP[TM * LSP];
  __shared__ unsigned short lsH[TK * LSB];

  const int tid  = threadIdx.x;
  const int bm   = blockIdx.y * TM;
  const int bn   = blockIdx.x * TN;
  const int wid  = tid >> 5, lane = tid & 31;
  const int wm   = wid & 3,  wn   = wid >> 2;
  const int ml   = lane & 15, hf  = lane >> 4;
  const int nb0  = wn * 32;

  const bool fullM = (bm + TM <= n);
  const unsigned ldsH = (unsigned)(unsigned long long)(const void*)&lsH[0];

  v8f acc0 = {0.f, 0.f, 0.f, 0.f, 0.f, 0.f, 0.f, 0.f};
  v8f acc1 = {0.f, 0.f, 0.f, 0.f, 0.f, 0.f, 0.f, 0.f};

  const int Kt = (n + TK - 1) / TK;
  for (int kt = 0; kt < Kt; ++kt) {
    const int k0 = kt * TK;
    const bool fullK = (k0 + TK <= n);

    // ---- H tile 32 x 64 bf16 via TDM (nc == 320 => bn+64 <= nc) ----
    if (fullK) {
      if (wid == 0)
        tdm_load_2d(ldsH, &H[(size_t)k0 * nc + bn], /*dsz=*/1u,
                    (unsigned)nc, (unsigned)n, TN, TK, (unsigned long long)nc);
    } else {
#pragma unroll
      for (int t = 0; t < 8; ++t) {
        int e = tid + t * 256;
        int r = e >> 6, c = e & 63;
        int gr = k0 + r, gc = bn + c;
        bool in = gr < n;
        unsigned short x = H[in ? ((size_t)gr * nc + gc) : 0];
        lsH[r * LSB + c] = in ? x : (unsigned short)0;
      }
    }
    // ---- P tile 64x32: masked, normalized attention weights ----
    if (fullM && fullK) {                      // adjacency lda always 4-aligned
#pragma unroll
      for (int t = 0; t < 2; ++t) {
        int e = tid + t * 256;
        int r = e >> 3, c4 = (e & 7) * 4;
        int gr = bm + r, gc = k0 + c4;
        const float4 a = *(const float4*)&Adj[(size_t)gr * lda + gc];
        const float4 edv = *(const float4*)&ed[gc];
        float esi = es[gr], mi = rowm[gr], si = rsinv[gr];
        float p[4];
        float av[4] = {a.x, a.y, a.z, a.w};
        float ev[4] = {edv.x, edv.y, edv.z, edv.w};
#pragma unroll
        for (int j = 0; j < 4; ++j) {
          float x = esi + ev[j];
          x = x >= 0.f ? x : 0.2f * x;                 // leaky_relu(0.2)
          float w = __expf(x - mi) * si;
          p[j] = av[j] > 0.f ? w : 0.f;
        }
        *(uint2*)&lsP[r * LSP + c4] = make_uint2(pk2bf(p[0], p[1]), pk2bf(p[2], p[3]));
      }
    } else {
#pragma unroll
      for (int t = 0; t < 4; ++t) {
        int e = tid + t * 256;
        int r = e >> 4, c2 = (e & 15) * 2;
        int gr = bm + r, gc = k0 + c2;
        bool rin = gr < n;
        float esi = es[rin ? gr : 0], mi = rowm[rin ? gr : 0], si = rsinv[rin ? gr : 0];
        float p[2];
#pragma unroll
        for (int j = 0; j < 2; ++j) {
          bool in = rin & (gc + j < n);
          float a  = Adj[in ? ((size_t)gr * lda + gc + j) : 0];
          float ev = ed[in ? (gc + j) : 0];
          float x = esi + ev;
          x = x >= 0.f ? x : 0.2f * x;
          float w = __expf(x - mi) * si;
          p[j] = (in && a > 0.f) ? w : 0.f;
        }
        *(unsigned*)&lsP[r * LSP + c2] = pk2bf(p[0], p[1]);
      }
    }
    __builtin_amdgcn_s_wait_tensorcnt((short)0);
    __syncthreads();

    FragBF fa, fb0, fb1;
#pragma unroll
    for (int q = 0; q < 8; ++q) {
      int kk = (q < 4) ? (2 * q + 8 * hf) : (16 + 2 * (q - 4) + 8 * hf);
      fa.u[q] = *(const unsigned*)&lsP[(wm * 16 + ml) * LSP + kk];
      unsigned lo0 = lsH[kk * LSB + nb0 + ml];
      unsigned hi0 = lsH[(kk + 1) * LSB + nb0 + ml];
      fb0.u[q] = lo0 | (hi0 << 16);
      unsigned lo1 = lsH[kk * LSB + nb0 + 16 + ml];
      unsigned hi1 = lsH[(kk + 1) * LSB + nb0 + 16 + ml];
      fb1.u[q] = lo1 | (hi1 << 16);
    }
    acc0 = __builtin_amdgcn_wmma_f32_16x16x32_bf16(false, fa.v, false, fb0.v,
                                                   (short)0, acc0, false, false);
    acc1 = __builtin_amdgcn_wmma_f32_16x16x32_bf16(false, fa.v, false, fb1.v,
                                                   (short)0, acc1, false, false);
    __syncthreads();
  }

#pragma unroll
  for (int q = 0; q < 8; ++q) {
    int row = bm + wm * 16 + q + 8 * hf;
    if (row >= n) continue;
#pragma unroll
    for (int s = 0; s < 2; ++s) {
      int col = bn + nb0 + s * 16 + ml;
      if (col < nc) {
        float x = (s == 0) ? acc0[q] : acc1[q];
        x = x > 0.f ? x : (__expf(x) - 1.f);             // ELU
        O[(size_t)row * nc + col] = x;
      }
    }
  }
}

// f32 -> bf16 bulk conversion (vectorized), n assumed >= 4
__global__ void cvt_bf16(const float* __restrict__ s,
                         unsigned short* __restrict__ d, long long n) {
  long long i = ((long long)blockIdx.x * blockDim.x + threadIdx.x) * 4;
  if (i + 3 < n) {
    float4 f = *(const float4*)&s[i];
    *(uint2*)&d[i] = make_uint2(pk2bf(f.x, f.y), pk2bf(f.z, f.w));
  } else {
    for (long long j = i; j < n; ++j) d[j] = f2bf(s[j]);
  }
}

// ---------------------------------------------------------------------------
// Per-row masked-softmax stats (vectorized float4 adjacency reads).
// ---------------------------------------------------------------------------
__global__ __launch_bounds__(256)
void gat_stats(const float* __restrict__ Adj, int lda, int n,
               const float* __restrict__ es, const float* __restrict__ ed,
               float* __restrict__ rowm, float* __restrict__ rsinv) {
  int row  = blockIdx.x * 8 + (threadIdx.x >> 5);
  int lane = threadIdx.x & 31;
  if (row >= n) return;
  const float* Ar = Adj + (size_t)row * lda;
  float esi = es[row];
  int n4 = n & ~127;                      // 32 lanes * 4
  float mx = -3.0e38f;
  for (int j0 = lane * 4; j0 < n4; j0 += 128) {
    float4 a = *(const float4*)&Ar[j0];
    float4 e = *(const float4*)&ed[j0];
    float av[4] = {a.x, a.y, a.z, a.w};
    float ev[4] = {e.x, e.y, e.z, e.w};
#pragma unroll
    for (int j = 0; j < 4; ++j) {
      float x = esi + ev[j];
      x = x >= 0.f ? x : 0.2f * x;
      mx = av[j] > 0.f ? fmaxf(mx, x) : mx;
    }
  }
  for (int j = n4 + lane; j < n; j += 32) {
    if (Ar[j] > 0.f) {
      float x = esi + ed[j];
      x = x >= 0.f ? x : 0.2f * x;
      mx = fmaxf(mx, x);
    }
  }
  for (int o = 16; o; o >>= 1) mx = fmaxf(mx, __shfl_xor(mx, o, 32));
  float sum = 0.f;
  for (int j0 = lane * 4; j0 < n4; j0 += 128) {
    float4 a = *(const float4*)&Ar[j0];
    float4 e = *(const float4*)&ed[j0];
    float av[4] = {a.x, a.y, a.z, a.w};
    float ev[4] = {e.x, e.y, e.z, e.w};
#pragma unroll
    for (int j = 0; j < 4; ++j) {
      float x = esi + ev[j];
      x = x >= 0.f ? x : 0.2f * x;
      sum += av[j] > 0.f ? __expf(x - mx) : 0.f;
    }
  }
  for (int j = n4 + lane; j < n; j += 32) {
    if (Ar[j] > 0.f) {
      float x = esi + ed[j];
      x = x >= 0.f ? x : 0.2f * x;
      sum += __expf(x - mx);
    }
  }
  for (int o = 16; o; o >>= 1) sum += __shfl_xor(sum, o, 32);
  if (lane == 0) { rowm[row] = mx; rsinv[row] = 1.f / sum; }
}

// es[i] = H[i,:].a[0:d], ed[i] = H[i,:].a[d:2d]   (one wave per row)
__global__ __launch_bounds__(256)
void matvec2(const float* __restrict__ H, int d, const float* __restrict__ a,
             float* __restrict__ es, float* __restrict__ ed, int n) {
  int row  = blockIdx.x * 8 + (threadIdx.x >> 5);
  int lane = threadIdx.x & 31;
  if (row >= n) return;
  const float* h = H + (size_t)row * d;
  float s0 = 0.f, s1 = 0.f;
  for (int j = lane; j < d; j += 32) { float hv = h[j]; s0 += hv * a[j]; s1 += hv * a[d + j]; }
  for (int o = 16; o; o >>= 1) { s0 += __shfl_xor(s0, o, 32); s1 += __shfl_xor(s1, o, 32); }
  if (lane == 0) { es[row] = s0; ed[row] = s1; }
}

// s[i] = sigmoid((X[i,:].w + b)/100)
__global__ __launch_bounds__(256)
void pool_scores(const float* __restrict__ X, const float* __restrict__ w,
                 const float* __restrict__ b, float* __restrict__ s, int n, int d) {
  int row  = blockIdx.x * 8 + (threadIdx.x >> 5);
  int lane = threadIdx.x & 31;
  if (row >= n) return;
  const float* x = X + (size_t)row * d;
  float acc = 0.f;
  for (int j = lane; j < d; j += 32) acc += x[j] * w[j];
  for (int o = 16; o; o >>= 1) acc += __shfl_xor(acc, o, 32);
  if (lane == 0) {
    float z = (acc + b[0]) * 0.01f;
    s[row] = 1.f / (1.f + __expf(-z));
  }
}

// deterministic stable top-k by rank counting (matches jax.lax.top_k ordering)
__global__ void topk_select(const float* __restrict__ s, int n, int kn,
                            int* __restrict__ idx) {
  int i = blockIdx.x * blockDim.x + threadIdx.x;
  if (i >= n) return;
  float si = s[i];
  int rank = 0;
  for (int j = 0; j < n; ++j) {
    float sj = s[j];
    rank += (sj > si) || (sj == si && j < i);
  }
  if (rank < kn) idx[rank] = i;
}

// Xo[r,:] = X[idx[r],:] * s[idx[r]]
__global__ void gather_rows_scale(const float* __restrict__ X, int ldx,
                                  const float* __restrict__ s,
                                  const int* __restrict__ idx,
                                  float* __restrict__ Xo, int kn, int d) {
  int r = blockIdx.y;
  int c = blockIdx.x * blockDim.x + threadIdx.x;
  if (r >= kn || c >= d) return;
  int src = idx[r];
  Xo[(size_t)r * d + c] = X[(size_t)src * ldx + c] * s[src];
}

// Ao[r,c] = A[idx[r], idx[c]]   (Ao has padded row stride ldo)
__global__ void gather_adj(const float* __restrict__ A, int ldbig,
                           const int* __restrict__ idx,
                           float* __restrict__ Ao, int kn, int ldo) {
  int r = blockIdx.y;
  int c = blockIdx.x * blockDim.x + threadIdx.x;
  if (r >= kn || c >= kn) return;
  Ao[(size_t)r * ldo + c] = A[(size_t)idx[r] * ldbig + idx[c]];
}

__global__ void zero_f(float* __restrict__ p, long long n) {
  long long i = (long long)blockIdx.x * blockDim.x + threadIdx.x;
  if (i < n) p[i] = 0.f;
}

// Xb[idx[r],:] = Xs[r,:]
__global__ void scatter_rows(const float* __restrict__ Xs,
                             const int* __restrict__ idx,
                             float* __restrict__ Xb, int kn, int d) {
  int r = blockIdx.y;
  int c = blockIdx.x * blockDim.x + threadIdx.x;
  if (r >= kn || c >= d) return;
  Xb[(size_t)idx[r] * d + c] = Xs[(size_t)r * d + c];
}

// O[i, 0:d] = X1[i,:], O[i, d:2d] = X2[i,:]
__global__ void concat_cols(const float* __restrict__ X1,
                            const float* __restrict__ X2,
                            float* __restrict__ O, int n, int d) {
  int i = blockIdx.y;
  int c = blockIdx.x * blockDim.x + threadIdx.x;
  if (i >= n || c >= 2 * d) return;
  O[(size_t)i * 2 * d + c] = (c < d) ? X1[(size_t)i * d + c]
                                     : X2[(size_t)i * d + (c - d)];
}

__global__ void copy_f(const float* __restrict__ src, float* __restrict__ dst,
                       long long n) {
  long long i = (long long)blockIdx.x * blockDim.x + threadIdx.x;
  if (i < n) dst[i] = src[i];
}

// ---------------------------------------------------------------------------
// Host orchestration
// ---------------------------------------------------------------------------
extern "C" void kernel_launch(void* const* d_in, const int* in_sizes, int n_in,
                              void* d_out, int out_size, void* d_ws, size_t ws_size,
                              hipStream_t stream) {
  (void)in_sizes; (void)n_in; (void)out_size; (void)ws_size;
  constexpr int N0 = 6144, N1 = 4915, N2 = 2949, D = 320;
  constexpr int LD1 = 4916, LD2 = 2952;   // 4-aligned row strides for pooled adj

  const float* A0   = (const float*)d_in[0];
  const float* X0   = (const float*)d_in[1];
  const float* sgW  = (const float*)d_in[2];
  const float* sga  = (const float*)d_in[3];
  const float* bgW  = (const float*)d_in[4];
  const float* bga  = (const float*)d_in[5];
  const float* egW  = (const float*)d_in[6];
  const float* ega  = (const float*)d_in[7];
  const float* dW   = (const float*)d_in[8];
  const float* db   = (const float*)d_in[9];
  const float* uW   = (const float*)d_in[10];
  const float* ub   = (const float*)d_in[11];
  const float* pw   = (const float*)d_in[12];
  const float* pb   = (const float*)d_in[13];

  // workspace layout (floats, 16B-aligned slots); ~301 MB total
  float* wsp = (float*)d_ws;
  size_t off = 0;
  auto alloc = [&](size_t cnt) {
    cnt = (cnt + 3) & ~(size_t)3;
    float* p = wsp + off; off += cnt; return p;
  };
  float* orgX  = alloc((size_t)N0 * D);   // = "start" output
  float* down0 = alloc((size_t)N0 * D);
  float* down1 = alloc((size_t)N1 * D);
  float* Xh    = alloc((size_t)N0 * D);   // GAT hidden h
  float* Xt    = alloc((size_t)N0 * D);   // GCN temp A@X
  float* Xa    = alloc((size_t)N0 * D);
  float* Xb    = alloc((size_t)N0 * D);
  float* Xcat  = alloc((size_t)N0 * 2 * D);
  unsigned short* bfB = (unsigned short*)alloc(((size_t)N0 * D + 1) / 2);
  float* es    = alloc(N0);
  float* ed    = alloc(N0);
  float* rm    = alloc(N0);
  float* rs    = alloc(N0);
  float* sc    = alloc(N0);
  int*   idx0  = (int*)alloc(N1);
  int*   idx1  = (int*)alloc(N2);
  int*   idx2  = (int*)alloc(N1);
  int*   idx3  = (int*)alloc(N2);
  float* A1    = alloc((size_t)N1 * LD1); // pass-1 A1 (reused by pass-2 up loop)
  float* A1b   = alloc((size_t)N1 * LD1); // pass-2 A1
  float* A2    = alloc((size_t)N2 * LD2); // shared by both passes (dead between)

  auto cvt = [&](const float* src, long long n) {
    cvt_bf16<<<(unsigned)((n / 4 + 255) / 256), 256, 0, stream>>>(src, bfB, n);
  };
  auto gemm = [&](const float* Am, int lda, int ldb,
                  float* Cm, int ldc, int M, int Nn, int K,
                  const float* bias, const float* resid, int act) {
    dim3 g((Nn + TN - 1) / TN, (M + TM - 1) / TM);
    wmma_gemm_bf16<<<g, dim3(256), 0, stream>>>(Am, lda, bfB, ldb, Cm, ldc,
                                                M, Nn, K, bias, resid, act);
  };
  auto gat = [&](const float* Adj, int ldadj, int n, const float* Xin, int ldx,
                 int K, const float* W, const float* avec, float* h, float* out) {
    cvt(W, (long long)K * D);
    gemm(Xin, ldx, D, h, D, n, D, K, nullptr, nullptr, 0);         // h = Xin@W
    matvec2<<<(n + 7) / 8, 256, 0, stream>>>(h, D, avec, es, ed, n);
    gat_stats<<<(n + 7) / 8, 256, 0, stream>>>(Adj, ldadj, n, es, ed, rm, rs);
    cvt(h, (long long)n * D);
    dim3 g((D + TN - 1) / TN, (n + TM - 1) / TM);
    wmma_gat_av<<<g, dim3(256), 0, stream>>>(Adj, ldadj, n, es, ed, rm, rs,
                                             bfB, out, D);
  };
  auto gcn = [&](const float* Adj, int ldadj, int n, const float* Xin,
                 const float* W, const float* b, const float* resid, float* out) {
    cvt(Xin, (long long)n * D);
    gemm(Adj, ldadj, D, Xt, D, n, D, n, nullptr, nullptr, 0);      // T = A@X
    cvt(W, (long long)D * D);
    gemm(Xt, D, D, out, D, n, D, D, b, resid, 0);                  // T@W + b (+R)
  };
  auto pool = [&](const float* Adj, int ldadj, int n, const float* Xin, int kn,
                  const float* w1, const float* b1, int* idx, float* Xs,
                  float* Asm, int ldo) {
    pool_scores<<<(n + 7) / 8, 256, 0, stream>>>(Xin, w1, b1, sc, n, D);
    topk_select<<<(n + 255) / 256, 256, 0, stream>>>(sc, n, kn, idx);
    gather_rows_scale<<<dim3((D + 255) / 256, kn), 256, 0, stream>>>(
        Xin, D, sc, idx, Xs, kn, D);
    gather_adj<<<dim3((kn + 255) / 256, kn), 256, 0, stream>>>(Adj, ldadj, idx,
                                                               Asm, kn, ldo);
  };
  auto unpool = [&](const float* Xs, int kn, const int* idx, float* Xbig, int nbig) {
    long long tot = (long long)nbig * D;
    zero_f<<<(unsigned)((tot + 255) / 256), 256, 0, stream>>>(Xbig, tot);
    scatter_rows<<<dim3((D + 255) / 256, kn), 256, 0, stream>>>(Xs, idx, Xbig, kn, D);
  };

  // ---- initial sg GAT: X = gat(A0, X0);  start = org_X = this X -------------
  gat(A0, N0, N0, X0, D, D, sgW, sga, Xh, orgX);

  // ---- two U-Net passes (note: up-loop always uses PASS-1 A1/idx/down_outs) -
  auto run_pass = [&](const float* Xin, int passid) {
    float* X1  = (passid == 0) ? down0 : Xa;
    gcn(A0, N0, N0, Xin, dW, db, nullptr, X1);
    int*   ia  = (passid == 0) ? idx0 : idx2;
    float* A1c = (passid == 0) ? A1 : A1b;
    pool(A0, N0, N0, X1, N1, pw, pb, ia, Xb, A1c, LD1);     // Xb = pooled X [N1,D]
    float* X2  = (passid == 0) ? down1 : Xa;
    gcn(A1c, LD1, N1, Xb, dW + D * D, db + D, nullptr, X2);
    int*   ib  = (passid == 0) ? idx1 : idx3;
    pool(A1c, LD1, N1, X2, N2, pw + D, pb + 1, ib, Xb, A2, LD2);  // [N2,D]
    // bottom GAT
    gat(A2, LD2, N2, Xb, D, D, bgW, bga, Xh, Xa);           // Xa = bottom out
    // up i=0: adj_ms[1] = pass-1 A1, idx_list[1] = pass-1 idx1, down_outs[1]
    unpool(Xa, N2, idx1, Xb, N1);
    gcn(A1, LD1, N1, Xb, uW, ub, down1, Xa);
    // up i=1: adj_ms[0] = A0, idx0, down_outs[0]
    unpool(Xa, N1, idx0, Xb, N0);
    gcn(A0, N0, N0, Xb, uW + D * D, ub + D, down0, Xa);
    // concat [X, org_X] -> [N0, 2D], then eg GAT (K = 2D)
    concat_cols<<<dim3((2 * D + 255) / 256, N0), 256, 0, stream>>>(Xa, orgX, Xcat, N0, D);
    gat(A0, N0, N0, Xcat, 2 * D, 2 * D, egW, ega, Xh, Xb);  // result in Xb
  };

  run_pass(orgX, 0);
  run_pass(Xb, 1);

  // ---- outputs: (X_final, start) concatenated flat ----
  long long tot = (long long)N0 * D;
  copy_f<<<(unsigned)((tot + 255) / 256), 256, 0, stream>>>(Xb, (float*)d_out, tot);
  copy_f<<<(unsigned)((tot + 255) / 256), 256, 0, stream>>>(orgX, (float*)d_out + tot, tot);
}
